// LstmNet_4544075399852
// MI455X (gfx1250) — matrix operand order repaired and checked by reading, not compiled, not run on hardware
//
#include <hip/hip_runtime.h>
#include <hip/hip_bf16.h>

// ---------------------------------------------------------------------------
// LSTM net: proj(leaky_relu) -> 5x LSTM(H=1024) -> leaky_relu -> proj(tanh)
// SEQ=8192, B=1.  Strategy:
//  * All x@Wih^T work hoisted out of the recurrence into bf16 WMMA GEMMs
//    (v_wmma_f32_16x16x32_bf16), f32 accumulate, bias fused in epilogue.
//  * Sequential h@Whh^T recurrence: persistent 64-WG kernel, fp32 VALU GEMV,
//    Whh stays L2-resident (84MB fp32 < 192MB L2), grid barrier per step.
// ---------------------------------------------------------------------------

#define SEQ   8192
#define NIN   64
#define HDIM  1024
#define G4H   (4 * HDIM)
#define NLAY  5
#define NWG   64          // workgroups in persistent recurrence kernel
#define JPW   (HDIM / NWG) // hidden units per workgroup = 16

typedef __attribute__((ext_vector_type(16))) __bf16        v16bf;
typedef __attribute__((ext_vector_type(8)))  float         v8f;
typedef __attribute__((ext_vector_type(4)))  unsigned int  v4u;
typedef __attribute__((ext_vector_type(8)))  unsigned int  v8u;

union FragA { v16bf v; v4u h[2]; };
union FragB { v16bf v; v8u u; };

// ---------------------------------------------------------------------------
__global__ void init_counter(unsigned int* c) { *c = 0u; }

__global__ void f32_to_bf16_kernel(const float* __restrict__ in,
                                   __hip_bfloat16* __restrict__ out, size_t n) {
    size_t i = (size_t)blockIdx.x * blockDim.x + threadIdx.x;
    if (i < n) out[i] = __float2bfloat16(in[i]);
}

// ---------------------------------------------------------------------------
// C(MxN) = A(MxK) @ W(NxK)^T + bias0 + bias1, optional leaky_relu, stores
// f32 and/or bf16.  Block = 256 threads = 8 waves tiled 2(M) x 4(N); each
// wave owns a 16x64 output tile (4 accumulators), K-loop step 32.
// A/B fragments use contiguous k = 16*lh + s (same order in both); C/D per ISA (wave32).
// ---------------------------------------------------------------------------
__global__ void wmma_gemm_bf16(const __hip_bfloat16* __restrict__ A,
                               const __hip_bfloat16* __restrict__ W,
                               const float* __restrict__ bias0,
                               const float* __restrict__ bias1,
                               float* __restrict__ Cf32,
                               __hip_bfloat16* __restrict__ Cbf16,
                               int M, int N, int K, int act) {
    const int lane = threadIdx.x & 31;
    const int wave = threadIdx.x >> 5;
    const int wm   = wave & 1;
    const int wn   = wave >> 1;
    const int m0   = (blockIdx.x * 2 + wm) * 16;
    const int n0   = (blockIdx.y * 4 + wn) * 64;
    const int lm   = lane & 15;   // A row / B column / C column within tile
    const int lh   = lane >> 4;   // K-half selector (A,B) / M-half (C)

    v8f acc[4] = {v8f{0.f,0.f,0.f,0.f,0.f,0.f,0.f,0.f},
                  v8f{0.f,0.f,0.f,0.f,0.f,0.f,0.f,0.f},
                  v8f{0.f,0.f,0.f,0.f,0.f,0.f,0.f,0.f},
                  v8f{0.f,0.f,0.f,0.f,0.f,0.f,0.f,0.f}};

    const __hip_bfloat16* arow = A + (size_t)(m0 + lm) * K;

    for (int k0 = 0; k0 < K; k0 += 32) {
        // A 16x32 bf16 frag: lane holds row lm; lanes 0-15 K={0..15},
        // lanes 16-31 K={16..31} -> two contiguous 16B chunks.
        FragA a;
        a.h[0] = *(const v4u*)(arow + k0 + lh * 16);
        a.h[1] = *(const v4u*)(arow + k0 + lh * 16 + 8);
#pragma unroll
        for (int nb = 0; nb < 4; ++nb) {
            // B 32x16 bf16 frag: lane = column (lm), 16 consecutive K values
            // (lh selects K 0-15 vs 16-31).  B[k][n] = W[n][k] -> contiguous
            // 32B read from row (n0+nb*16+lm) of W.
            FragB b;
            b.u = *(const v8u*)(W + (size_t)(n0 + nb * 16 + lm) * K + k0 + lh * 16);
            acc[nb] = __builtin_amdgcn_wmma_f32_16x16x32_bf16(
                false, a.v, false, b.v, (short)0, acc[nb], false, false);
        }
    }

    // Epilogue. C/D layout: VGPR r -> row m0 + r + 8*lh, column n0 + ... + lm.
#pragma unroll
    for (int nb = 0; nb < 4; ++nb) {
        const int n = n0 + nb * 16 + lm;
        float bsum = 0.f;
        if (bias0) bsum += bias0[n];
        if (bias1) bsum += bias1[n];
#pragma unroll
        for (int r = 0; r < 8; ++r) {
            const int m = m0 + r + 8 * lh;
            float v = acc[nb][r] + bsum;
            if (act == 1) v = (v >= 0.f) ? v : 0.01f * v;
            if (Cf32)  Cf32[(size_t)m * N + n] = v;
            if (Cbf16) Cbf16[(size_t)m * N + n] = __float2bfloat16(v);
        }
    }
}

// ---------------------------------------------------------------------------
__device__ __forceinline__ float sigmoidf_(float x) {
    return 1.0f / (1.0f + __expf(-x));
}

__device__ __forceinline__ void grid_barrier(unsigned int* counter,
                                             unsigned int target, int tid) {
    __threadfence();          // make h writes device-visible before arrival
    __syncthreads();
    if (tid == 0) {
        atomicAdd(counter, 1u);
        while (*(volatile unsigned int*)counter < target)
            __builtin_amdgcn_s_sleep(2);
    }
    __syncthreads();
    __threadfence();
}

// ---------------------------------------------------------------------------
// Persistent recurrence: 64 WGs x 256 threads.  WG `wg` owns hidden units
// j in [wg*16, wg*16+16): computes gate rows {j, H+j, 2H+j, 3H+j} (so the
// c/h update is WG-local), keeps c in registers, ping-pongs h through a
// global buffer with one grid barrier per timestep.
// ---------------------------------------------------------------------------
__global__ void lstm_recurrence(const float* __restrict__ Zx,   // (SEQ,4H) incl. bias
                                const float* __restrict__ Whh,  // (4H,H) fp32
                                float* __restrict__ hping,      // 2*H
                                __hip_bfloat16* __restrict__ xnext_bf16, // (SEQ,H)
                                float* __restrict__ xnext_f32,  // (SEQ,H)
                                const float* __restrict__ h0,
                                const float* __restrict__ c0,
                                unsigned int* counter, int layer) {
    __shared__ float hs[HDIM];
    __shared__ float partial[64][4];
    __shared__ float zg[4][JPW];

    const int tid = threadIdx.x;
    const int wg  = blockIdx.x;
    const int j0  = wg * JPW;
    const int d   = tid >> 2;        // dot index 0..63 (4 gates x 16 units)
    const int p   = tid & 3;         // K-partition 0..3 (256 elems each)
    const int g   = d >> 4;          // gate: 0=i 1=f 2=g 3=o
    const int jj  = d & 15;
    const int row = g * HDIM + j0 + jj;
    const float4* wrow4 = (const float4*)(Whh + (size_t)row * HDIM + p * (HDIM / 4));

    float cst = 0.f;
    if (tid < JPW) {
        cst = c0[j0 + tid];
        hping[j0 + tid] = h0[j0 + tid];   // ping slot 0
    }

    const unsigned int base = (unsigned int)layer * (SEQ + 1) * NWG;
    grid_barrier(counter, base + NWG, tid);

    for (int t = 0; t < SEQ; ++t) {
        const float* hsrc = hping + (t & 1) * HDIM;
        for (int k = tid; k < HDIM; k += 256) hs[k] = hsrc[k];
        __syncthreads();

        // 1024-length dot, 4 threads per dot, float4 streams from L2.
        float acc = 0.f;
        const float4* h4 = (const float4*)(hs + p * (HDIM / 4));
#pragma unroll 4
        for (int k = 0; k < HDIM / 16; ++k) {
            float4 w = wrow4[k];
            float4 h = h4[k];
            acc = fmaf(w.x, h.x, acc);
            acc = fmaf(w.y, h.y, acc);
            acc = fmaf(w.z, h.z, acc);
            acc = fmaf(w.w, h.w, acc);
        }
        partial[d][p] = acc;
        __syncthreads();

        if (p == 0) {
            float z = partial[d][0] + partial[d][1] + partial[d][2] + partial[d][3]
                    + Zx[(size_t)t * G4H + row];
            zg[g][jj] = z;
        }
        __syncthreads();

        if (tid < JPW) {
            const int j = j0 + tid;
            const float i = sigmoidf_(zg[0][tid]);
            const float f = sigmoidf_(zg[1][tid]);
            const float gg = tanhf(zg[2][tid]);
            const float o = sigmoidf_(zg[3][tid]);
            cst = f * cst + i * gg;
            const float h = o * tanhf(cst);
            hping[((t + 1) & 1) * HDIM + j] = h;
            xnext_f32[(size_t)t * HDIM + j] = h;
            xnext_bf16[(size_t)t * HDIM + j] = __float2bfloat16(h);
        }
        grid_barrier(counter, base + (unsigned int)(t + 2) * NWG, tid);
    }
}

// ---------------------------------------------------------------------------
// out[t][n] = tanh(b2[n] + sum_k leaky_relu(X[t][k]) * w2[n][k])
// One block per timestep; row staged through LDS with activation applied.
// ---------------------------------------------------------------------------
__global__ void final_proj(const float* __restrict__ X,
                           const float* __restrict__ w2,
                           const float* __restrict__ b2,
                           float* __restrict__ out) {
    __shared__ float xs[HDIM];
    const int t = blockIdx.x;
    for (int k = threadIdx.x; k < HDIM; k += blockDim.x) {
        float v = X[(size_t)t * HDIM + k];
        xs[k] = (v >= 0.f) ? v : 0.01f * v;
    }
    __syncthreads();
    const int n = threadIdx.x;
    if (n < NIN) {
        const float4* w4 = (const float4*)(w2 + (size_t)n * HDIM);
        const float4* x4 = (const float4*)xs;
        float acc = b2[n];
#pragma unroll 4
        for (int k = 0; k < HDIM / 4; ++k) {
            float4 w = w4[k];
            float4 x = x4[k];
            acc = fmaf(w.x, x.x, acc);
            acc = fmaf(w.y, x.y, acc);
            acc = fmaf(w.z, x.z, acc);
            acc = fmaf(w.w, x.w, acc);
        }
        out[(size_t)t * NIN + n] = tanhf(acc);
    }
}

// ---------------------------------------------------------------------------
extern "C" void kernel_launch(void* const* d_in, const int* in_sizes, int n_in,
                              void* d_out, int out_size, void* d_ws, size_t ws_size,
                              hipStream_t stream) {
    (void)in_sizes; (void)n_in; (void)out_size; (void)ws_size;

    const float* data_in = (const float*)d_in[0];   // (SEQ,1,NIN)
    const float* w1      = (const float*)d_in[1];   // (H,NIN)
    const float* b1      = (const float*)d_in[2];   // (H)
    const float* W_ih    = (const float*)d_in[3];   // (L,4H,H)
    const float* W_hh    = (const float*)d_in[4];   // (L,4H,H)
    const float* b_ih    = (const float*)d_in[5];   // (L,4H)
    const float* b_hh    = (const float*)d_in[6];   // (L,4H)
    const float* w2      = (const float*)d_in[7];   // (NIN,H)
    const float* b2      = (const float*)d_in[8];   // (NIN)
    const float* h0      = (const float*)d_in[9];   // (L,1,H)
    const float* c0      = (const float*)d_in[10];  // (L,1,H)
    float* out = (float*)d_out;

    // ---- workspace carve-up --------------------------------------------
    char* ws = (char*)d_ws;
    size_t off = 0;
    auto carve = [&](size_t bytes) -> void* {
        void* p = ws + off;
        off = (off + bytes + 255) & ~(size_t)255;
        return p;
    };
    unsigned int*   counter   = (unsigned int*)  carve(256);
    __hip_bfloat16* din_bf    = (__hip_bfloat16*)carve((size_t)SEQ * NIN * 2);
    __hip_bfloat16* w1_bf     = (__hip_bfloat16*)carve((size_t)HDIM * NIN * 2);
    __hip_bfloat16* wih_bf    = (__hip_bfloat16*)carve((size_t)NLAY * G4H * HDIM * 2);
    __hip_bfloat16* xbuf0     = (__hip_bfloat16*)carve((size_t)SEQ * HDIM * 2);
    __hip_bfloat16* xbuf1     = (__hip_bfloat16*)carve((size_t)SEQ * HDIM * 2);
    float*          Zx        = (float*)         carve((size_t)SEQ * G4H * 4);
    float*          hping     = (float*)         carve((size_t)2 * HDIM * 4);
    float*          Xf32      = (float*)         carve((size_t)SEQ * HDIM * 4);
    __hip_bfloat16* xbufs[2]  = {xbuf0, xbuf1};

    // ---- barrier counter reset (fresh per launch, monotonic within) ----
    init_counter<<<1, 1, 0, stream>>>(counter);

    // ---- bf16 conversions (WMMA A/B operands) --------------------------
    {
        size_t n = (size_t)SEQ * NIN;
        f32_to_bf16_kernel<<<(unsigned)((n + 255) / 256), 256, 0, stream>>>(data_in, din_bf, n);
        n = (size_t)HDIM * NIN;
        f32_to_bf16_kernel<<<(unsigned)((n + 255) / 256), 256, 0, stream>>>(w1, w1_bf, n);
        n = (size_t)NLAY * G4H * HDIM;
        f32_to_bf16_kernel<<<(unsigned)((n + 255) / 256), 256, 0, stream>>>(W_ih, wih_bf, n);
    }

    // ---- input projection: X0 = leaky_relu(data_in @ w1^T + b1) --------
    {
        dim3 grid(SEQ / 32, HDIM / 256);
        wmma_gemm_bf16<<<grid, 256, 0, stream>>>(din_bf, w1_bf, b1, nullptr,
                                                 nullptr, xbufs[0],
                                                 SEQ, HDIM, NIN, /*act=*/1);
    }

    // ---- stacked LSTM layers -------------------------------------------
    for (int l = 0; l < NLAY; ++l) {
        // Zx = X_l @ W_ih[l]^T + (b_ih[l] + b_hh[l])   (WMMA bf16 GEMM)
        dim3 grid(SEQ / 32, G4H / 256);
        wmma_gemm_bf16<<<grid, 256, 0, stream>>>(
            xbufs[l & 1], wih_bf + (size_t)l * G4H * HDIM,
            b_ih + (size_t)l * G4H, b_hh + (size_t)l * G4H,
            Zx, nullptr, SEQ, G4H, HDIM, /*act=*/0);

        // sequential recurrence over 8192 steps (persistent, grid-synced)
        lstm_recurrence<<<NWG, 256, 0, stream>>>(
            Zx, W_hh + (size_t)l * G4H * HDIM, hping,
            xbufs[(l + 1) & 1], Xf32,
            h0 + (size_t)l * HDIM, c0 + (size_t)l * HDIM,
            counter, l);
    }

    // ---- output head: tanh(leaky_relu(X5) @ w2^T + b2) -----------------
    final_proj<<<SEQ, 64, 0, stream>>>(Xf32, w2, b2, out);
}
